// PxCL_56014963474805
// MI455X (gfx1250) — matrix operand order repaired and checked by reading, not compiled, not run on hardware
//
#include <hip/hip_runtime.h>
#include <hip/hip_bf16.h>
#include <math.h>

typedef __attribute__((ext_vector_type(16))) _Float16 v16h;
typedef __attribute__((ext_vector_type(8)))  float    v8f;

#define BATCH   2
#define CDIM    128
#define NPIX    65536        // H*W
#define CHUNKS  64
#define MBLK    1024         // NPIX/CHUNKS
#define LASTOFF (63 * 1024)  // start of last chunk
#define INV_T   (1.0f / 0.07f)

// ---------------- workspace layout ----------------
// A     : BATCH*NPIX*CDIM halves (normalized z1, row-major [b][n][c])
// Bfrag : BATCH*4*64*32*16 halves (packed WMMA-B fragments)
// pos   : BATCH*NPIX floats (already divided by T)
// rinv2 : BATCH*NPIX floats
// part  : 8192 floats (per-block partial losses)
#define A_HALVES   ((size_t)BATCH * NPIX * CDIM)          // 16777216
#define BF_HALVES  ((size_t)BATCH * 4 * 64 * 32 * 16)     // 262144
#define POS_FLOATS ((size_t)BATCH * NPIX)                 // 131072
#define NTILES     (BATCH * (NPIX / 16))                  // 8192

// ================= Kernel 1: norms, pos, normalized-A (f16) =================
__global__ void pxcl_norm_kernel(const float* __restrict__ z1,
                                 const float* __restrict__ z2,
                                 _Float16* __restrict__ A,
                                 float* __restrict__ pos,
                                 float* __restrict__ rinv2) {
  const int g = blockIdx.x * blockDim.x + threadIdx.x;   // b*NPIX + n
  const int b = g / NPIX;
  const int n = g % NPIX;
  const size_t base = (size_t)b * CDIM * NPIX + n;

  float s11 = 0.f, s22 = 0.f, s12 = 0.f;
#pragma unroll 4
  for (int c = 0; c < CDIM; ++c) {
    float a = z1[base + (size_t)c * NPIX];
    float d = z2[base + (size_t)c * NPIX];
    s11 += a * a; s22 += d * d; s12 += a * d;
  }
  const float i1 = 1.0f / fmaxf(sqrtf(s11), 1e-12f);
  const float i2 = 1.0f / fmaxf(sqrtf(s22), 1e-12f);

  pos[g]   = s12 * i1 * i2 * INV_T;   // logit 0, already /T
  rinv2[g] = i2;

  _Float16* arow = A + (size_t)g * CDIM;
#pragma unroll 4
  for (int c = 0; c < CDIM; ++c)
    arow[c] = (_Float16)(z1[base + (size_t)c * NPIX] * i1);
}

// ====== Kernel 1b: pack normalized z2 last chunk into WMMA B fragments ======
// B-matrix 32x16 f16 lane layout used here (contiguous k, same as A): lane<16 holds col=lane,
// K=e (e=0..15); lane>=16 holds col=lane-16, K=16+e.
__global__ void pxcl_packB_kernel(const float* __restrict__ z2,
                                  const float* __restrict__ rinv2,
                                  _Float16* __restrict__ Bfrag) {
  const int g = blockIdx.x * blockDim.x + threadIdx.x;   // [0, B*128*1024)
  const int b = g / (CDIM * MBLK);
  const int r = g % (CDIM * MBLK);
  const int k = r / MBLK;          // 0..127
  const int m = r % MBLK;          // 0..1023
  const int n = LASTOFF + m;

  const float v = z2[((size_t)b * CDIM + k) * NPIX + n] * rinv2[(size_t)b * NPIX + n];

  const int ks = k >> 5;           // k-step (32 per WMMA)
  const int kk = k & 31;
  const int ct = m >> 4;           // column tile 0..63
  const int lane = (m & 15) | ((kk >= 16) ? 16 : 0);
  const int e = kk & 15;
  Bfrag[((((size_t)b * 4 + ks) * 64 + ct) * 32 + lane) * 16 + e] = (_Float16)v;
}

// ============ Kernel 2: WMMA GEMM + fused masked log-softmax ================
__global__ void __launch_bounds__(256)
pxcl_gemm_lse_kernel(const _Float16* __restrict__ A,
                     const _Float16* __restrict__ Bfrag,
                     const float* __restrict__ pos,
                     const int* __restrict__ labels,
                     float* __restrict__ part) {
  __shared__ uint4  lds_a4[256];            // 16 rows x 128 k f16 (4 KB)
  __shared__ int    lds_lab[16];
  __shared__ float  lds_pos[16];
  __shared__ float  lds_red1[8 * 16];
  __shared__ float  lds_red2[8 * 16];
  __shared__ float  lds_gmax[16];
  __shared__ float  lds_loss[16];

  const int tid  = threadIdx.x;
  const int wave = tid >> 5;
  const int lane = tid & 31;
  const int tile = blockIdx.x;
  const int b       = tile / (NPIX / 16);
  const int rowbase = (tile % (NPIX / 16)) * 16;

  // ---- stage A tile (16 rows x 128 k, contiguous in ws) into LDS ----
  const uint4* asrc = (const uint4*)(A + ((size_t)b * NPIX + rowbase) * CDIM);
  lds_a4[tid] = asrc[tid];
  if (tid < 16) {
    lds_lab[tid] = labels[(size_t)b * NPIX + rowbase + tid];
    lds_pos[tid] = pos[(size_t)b * NPIX + rowbase + tid];
  }
  __syncthreads();

  const _Float16* lds_a = (const _Float16*)lds_a4;

  // ---- load A fragments in contiguous k order (same as packed B) ----
  // lane<16: row=lane, K=e (e=0..15)
  // lane>=16: row=lane-16, K=16+e (e=0..15)
  const int arow  = lane & 15;
  const int kb0   = (lane < 16) ? 0 : 16;
  const int kb1   = (lane < 16) ? 8 : 24;
  v16h af[4];
#pragma unroll
  for (int ks = 0; ks < 4; ++ks) {
    const _Float16* p0 = lds_a + arow * CDIM + ks * 32 + kb0;
    const _Float16* p1 = lds_a + arow * CDIM + ks * 32 + kb1;
#pragma unroll
    for (int e = 0; e < 8; ++e) { af[ks][e] = p0[e]; af[ks][8 + e] = p1[e]; }
  }

  // prefetch this wave's first B fragment (emits global_prefetch_b8)
  const _Float16* bwave0 =
      Bfrag + ((((size_t)b * 4) * 64 + wave * 8) * 32 + lane) * 16;
  __builtin_prefetch(bwave0, 0, 1);

  // ---- 8 column tiles x 4 k-steps of v_wmma_f32_16x16x32_f16 ----
  v8f acc[8];
#pragma unroll
  for (int ct = 0; ct < 8; ++ct) {
    v8f c = {};
#pragma unroll
    for (int ks = 0; ks < 4; ++ks) {
      const _Float16* bp =
          Bfrag + ((((size_t)b * 4 + ks) * 64 + (wave * 8 + ct)) * 32 + lane) * 16;
      v16h bb;
#pragma unroll
      for (int e = 0; e < 16; ++e) bb[e] = bp[e];
      c = __builtin_amdgcn_wmma_f32_16x16x32_f16(
          /*neg_a=*/false, af[ks], /*neg_b=*/false, bb,
          /*c_mod=*/(short)0, c, /*reuse_a=*/false, /*reuse_b=*/false);
    }
    acc[ct] = c;
  }

  // ---- mask (label inequality) + temperature, in place ----
  // C/D layout: lane<16 -> rows 0..7 (vgpr i), col=lane; lane>=16 -> rows 8..15, col=lane-16.
  const int rhalf = (lane < 16) ? 0 : 8;
  int labr[8];
#pragma unroll
  for (int i = 0; i < 8; ++i) labr[i] = lds_lab[rhalf + i];

#pragma unroll
  for (int ct = 0; ct < 8; ++ct) {
    const int col = (wave * 8 + ct) * 16 + (lane & 15);
    const int labc = labels[(size_t)b * NPIX + LASTOFF + col];
#pragma unroll
    for (int i = 0; i < 8; ++i)
      acc[ct][i] = (labr[i] != labc) ? acc[ct][i] * INV_T : 0.0f;
  }

  // ---- pass 1: per-row max over this lane's 8 columns, reduce ----
  float rmax[8];
#pragma unroll
  for (int i = 0; i < 8; ++i) {
    float m = acc[0][i];
#pragma unroll
    for (int ct = 1; ct < 8; ++ct) m = fmaxf(m, acc[ct][i]);
    for (int off = 1; off < 16; off <<= 1) m = fmaxf(m, __shfl_xor(m, off, 16));
    rmax[i] = m;
  }
  if ((lane & 15) == 0) {
#pragma unroll
    for (int i = 0; i < 8; ++i) lds_red1[wave * 16 + rhalf + i] = rmax[i];
  }
  __syncthreads();
  if (tid < 16) {
    float g = lds_pos[tid];
    for (int w = 0; w < 8; ++w) g = fmaxf(g, lds_red1[w * 16 + tid]);
    lds_gmax[tid] = g;
  }
  __syncthreads();

  // ---- pass 2: sum exp(v - gmax) and sum v ----
  float gm[8];
#pragma unroll
  for (int i = 0; i < 8; ++i) gm[i] = lds_gmax[rhalf + i];

  float se[8], sl[8];
#pragma unroll
  for (int i = 0; i < 8; ++i) {
    float e = 0.f, s = 0.f;
#pragma unroll
    for (int ct = 0; ct < 8; ++ct) {
      const float v = acc[ct][i];
      e += __expf(v - gm[i]);
      s += v;
    }
    for (int off = 1; off < 16; off <<= 1) {
      e += __shfl_xor(e, off, 16);
      s += __shfl_xor(s, off, 16);
    }
    se[i] = e; sl[i] = s;
  }
  if ((lane & 15) == 0) {
#pragma unroll
    for (int i = 0; i < 8; ++i) {
      lds_red1[wave * 16 + rhalf + i] = se[i];
      lds_red2[wave * 16 + rhalf + i] = sl[i];
    }
  }
  __syncthreads();

  // ---- finalize rows: loss_r = 1025*LSE_r - sum(logits_r) ----
  if (tid < 16) {
    float esum = __expf(lds_pos[tid] - lds_gmax[tid]);
    float lsum = lds_pos[tid];
    for (int w = 0; w < 8; ++w) { esum += lds_red1[w * 16 + tid]; lsum += lds_red2[w * 16 + tid]; }
    lds_loss[tid] = 1025.0f * (lds_gmax[tid] + __logf(esum)) - lsum;
  }
  __syncthreads();
  if (tid == 0) {
    float t = 0.f;
    for (int r = 0; r < 16; ++r) t += lds_loss[r];
    part[blockIdx.x] = t;
  }
}

// ================= Kernel 3: final reduction to scalar ======================
__global__ void pxcl_reduce_kernel(const float* __restrict__ part,
                                   float* __restrict__ out) {
  __shared__ float red[256];
  float s = 0.f;
  for (int i = threadIdx.x; i < NTILES; i += 256) s += part[i];
  red[threadIdx.x] = s;
  __syncthreads();
  for (int off = 128; off > 0; off >>= 1) {
    if (threadIdx.x < off) red[threadIdx.x] += red[threadIdx.x + off];
    __syncthreads();
  }
  if (threadIdx.x == 0)
    out[0] = red[0] * (1.0f / ((float)BATCH * CHUNKS * MBLK * 1025.0f));
}

// ============================== launcher ====================================
extern "C" void kernel_launch(void* const* d_in, const int* in_sizes, int n_in,
                              void* d_out, int out_size, void* d_ws, size_t ws_size,
                              hipStream_t stream) {
  const float* z1     = (const float*)d_in[0];
  const float* z2     = (const float*)d_in[1];
  const int*   labels = (const int*)d_in[2];
  (void)in_sizes; (void)n_in; (void)out_size; (void)ws_size;

  _Float16* A     = (_Float16*)d_ws;
  _Float16* Bfrag = A + A_HALVES;
  float*    pos   = (float*)(Bfrag + BF_HALVES);
  float*    rinv2 = pos + POS_FLOATS;
  float*    part  = rinv2 + POS_FLOATS;

  // K1: norms + pos + normalized A (f16)
  pxcl_norm_kernel<<<(BATCH * NPIX) / 256, 256, 0, stream>>>(z1, z2, A, pos, rinv2);
  // K1b: pack B fragments
  pxcl_packB_kernel<<<(BATCH * CDIM * MBLK) / 256, 256, 0, stream>>>(z2, rinv2, Bfrag);
  // K2: WMMA GEMM + fused masked log-softmax partials
  pxcl_gemm_lse_kernel<<<NTILES, 256, 0, stream>>>(A, Bfrag, pos, labels, part);
  // K3: scalar
  pxcl_reduce_kernel<<<1, 256, 0, stream>>>(part, (float*)d_out);
}
